// LogPolarFoveatedConvSNN_27616639713857
// MI455X (gfx1250) — matrix operand order repaired and checked by reading, not compiled, not run on hardware
//
#include <hip/hip_runtime.h>

typedef __attribute__((ext_vector_type(16))) _Float16 v16h;
typedef __attribute__((ext_vector_type(8)))  float    v8f;

#define T_STEPS 16
#define B_SZ    8
#define H_IMG   224
#define W_IMG   224
#define C_IN    3
#define RB      64
#define AB      128
#define C1      64
#define C2      128
#define OUT_DIM 1000

// ---- d_out float offsets (outputs concatenated flat in return order) ----
#define OFF_READOUT 0          // [8,1000]          8000
#define OFF_LOGITS  8000       // [16,8,1000]     128000
#define OFF_SR      136000     // [2]                  2
#define OFF_LP      136002     // [16,8,64,128,3] 3145728
#define OFF_RE      3281730    // [64]                64

// ---- workspace byte offsets ----
#define WS_V1    ((size_t)0)         // 4194304 f32  (B,RB,AB,C1) membrane 1
#define WS_V2    ((size_t)16777216)  // 8388608 f32  (B,RB,AB,C2) membrane 2
#define WS_S1    ((size_t)50331648)  // 4194304 f16  spikes layer 1
#define WS_W2P   ((size_t)58720256)  // 73728   f16  WMMA-fragment-packed w2
#define WS_S2SUM ((size_t)58867712)  // 16384   f32  [T,B,C2] spatial spike sums
#define WS_CNT   ((size_t)58933248)  // 16      f32  [0]=s1 spikes, [1]=s2 spikes

struct alignas(16) F4 { float x, y, z, w; };

// ---------------------------------------------------------------------------
// Zero all read-modify-write state every launch (determinism under graph replay)
// ---------------------------------------------------------------------------
__global__ __launch_bounds__(256) void init_kernel(float* v1, float* v2, float* s2sum,
                                                   float* cnt, float* out) {
  size_t i = (size_t)blockIdx.x * 256 + threadIdx.x;   // grid covers 8388608
  v2[i] = 0.f;
  if (i < 4194304) v1[i] = 0.f;
  if (i < 16384)   s2sum[i] = 0.f;
  if (i < 16)      cnt[i] = 0.f;
  if (i < 8000)    out[OFF_READOUT + i] = 0.f;
  if (i < 2)       out[OFF_SR + i] = 0.f;
  if (i < 64)      out[OFF_RE + i] = 0.f;
}

// ---------------------------------------------------------------------------
// Pack w2[3,3,64,128] f32 -> per-lane f16 B-fragments for v_wmma_f32_16x16x32_f16.
// Fragment id = (tap*2+kc)*8+nt, 512 halves each; lane L holds column N=L&15,
// K = (L>>4)*8 + e for e<8, 16 + (L>>4)*8 + (e-8) for e>=8 (matches A interleaved-K).
// ---------------------------------------------------------------------------
__global__ __launch_bounds__(256) void pack_w2_kernel(const float* w2, _Float16* w2p) {
  int i = blockIdx.x * 256 + threadIdx.x;
  if (i >= 9 * 2 * 8 * 512) return;
  int frag = i >> 9, within = i & 511;
  int lane = within >> 4, e = within & 15;
  int nt = frag & 7, kc = (frag >> 3) & 1, tap = frag >> 4;
  int lm = lane & 15, hi = lane >> 4;
  int ci = kc * 32 + (e < 8 ? hi * 8 + e : 16 + hi * 8 + (e - 8));
  int co = nt * 16 + lm;
  w2p[i] = (_Float16)w2[((size_t)tap * C1 + ci) * C2 + co];
}

// ---------------------------------------------------------------------------
// Log-polar bilinear resample for all T at once + radial-energy reduction.
// block = one (t,b,r) slice, 384 threads = (a, c).
// ---------------------------------------------------------------------------
__global__ __launch_bounds__(384) void sample_kernel(const float* x_seq, float* out) {
  int blk = blockIdx.x;              // (t*8+b)*64 + r
  int r = blk & 63;
  int tb = blk >> 6;
  int tid = threadIdx.x;
  int a = tid / 3, c = tid % 3;

  const float cy = (H_IMG - 1) * 0.5f;          // 111.5 ; log_min = log(1)=0
  const float log_max = logf(cy);
  float rad = expf(((float)r / (RB - 1)) * log_max);
  float ang = (float)a * (6.283185307179586f / AB);
  float ysf = fminf(fmaxf(cy + rad * sinf(ang), 0.f), (float)(H_IMG - 1));
  float xsf = fminf(fmaxf(cy + rad * cosf(ang), 0.f), (float)(W_IMG - 1));

  float y0 = floorf(ysf), x0 = floorf(xsf);
  float wy = ysf - y0, wx = xsf - x0;
  int y0i = (int)y0, x0i = (int)x0;
  int y1i = min(y0i + 1, H_IMG - 1), x1i = min(x0i + 1, W_IMG - 1);

  const float* img = x_seq + (size_t)tb * (H_IMG * W_IMG * C_IN);
  float g00 = img[((size_t)y0i * W_IMG + x0i) * C_IN + c];
  float g01 = img[((size_t)y0i * W_IMG + x1i) * C_IN + c];
  float g10 = img[((size_t)y1i * W_IMG + x0i) * C_IN + c];
  float g11 = img[((size_t)y1i * W_IMG + x1i) * C_IN + c];
  float v = g00 * (1.f - wy) * (1.f - wx) + g01 * (1.f - wy) * wx +
            g10 * wy * (1.f - wx) + g11 * wy * wx;

  out[OFF_LP + ((size_t)blk * AB + a) * C_IN + c] = v;

  __shared__ float sred;
  if (tid == 0) sred = 0.f;
  __syncthreads();
  atomicAdd(&sred, v);
  __syncthreads();
  if (tid == 0) atomicAdd(out + OFF_RE + r, sred * (1.0f / 6144.0f)); // /(T*AB*C)
}

// ---------------------------------------------------------------------------
// Conv1 (3x3, 3->64) + LIF(v1) -> f16 spikes. Small K (27): VALU fp32.
// block = 4 positions x 64 output channels.
// ---------------------------------------------------------------------------
__global__ __launch_bounds__(256) void conv1_lif_kernel(const float* out_lp, const float* w1,
                                                        const float* b1, float* v1,
                                                        _Float16* s1, float* cnt, int t) {
  int tid = threadIdx.x;
  int g = tid >> 6, co = tid & 63;
  int p = blockIdx.x * 4 + g;        // (b*RB+r)*AB + a
  int a = p & 127;
  int br = p >> 7;
  int r = br & 63, b = br >> 6;

  __shared__ float xs[4][28];
  __shared__ float sred;
  if (tid == 0) sred = 0.f;
  if (co < 27) {
    int tap = co / 3, ci = co % 3;
    int dr = tap / 3, da = tap % 3;
    int rin = r + dr - 1, ain = a + da - 1;
    float v = 0.f;
    if (rin >= 0 && rin < RB && ain >= 0 && ain < AB)
      v = out_lp[OFF_LP + (((size_t)(t * B_SZ + b) * RB + rin) * AB + ain) * C_IN + ci];
    xs[g][co] = v;
  }
  __syncthreads();

  float acc = b1[co];
#pragma unroll
  for (int j = 0; j < 27; ++j) acc += xs[g][j] * w1[j * C1 + co];

  size_t idx = (size_t)p * C1 + co;
  float v = 0.9f * v1[idx] + acc;
  float s = (v > 1.0f) ? 1.0f : 0.0f;
  v1[idx] = v - s;
  s1[idx] = (_Float16)s;

  atomicAdd(&sred, s);
  __syncthreads();
  if (tid == 0) atomicAdd(cnt + 0, sred);
}

// ---------------------------------------------------------------------------
// Conv2 (3x3, 64->128) as implicit GEMM on WMMA f16 + fused LIF(v2) + reductions.
// block = one (b,r) row: 8 waves x 16-wide a-tiles, each wave all 128 out ch.
// A = spikes from LDS (ISA 7.12.2 interleaved-K layout), B = packed weights.
// ---------------------------------------------------------------------------
__global__ __launch_bounds__(256) void conv2_lif_wmma_kernel(const _Float16* s1,
                                                             const _Float16* w2p,
                                                             const float* b2, float* v2,
                                                             float* s2sum, float* cnt, int t) {
  int tid = threadIdx.x;
  int b = blockIdx.x >> 6;
  int r = blockIdx.x & 63;

  __shared__ _Float16 tile[3][AB + 2][C1];   // rows r-1..r+1, a padded by 1 each side
  __shared__ float sred;
  if (tid == 0) sred = 0.f;

  for (int i = tid; i < 3 * (AB + 2); i += 256) {
    int rr = i / (AB + 2), aa = i % (AB + 2);
    int rin = r + rr - 1, ain = aa - 1;
    F4* dst = reinterpret_cast<F4*>(&tile[rr][aa][0]);
    if (rin < 0 || rin >= RB || ain < 0 || ain >= AB) {
      F4 z = {0.f, 0.f, 0.f, 0.f};
#pragma unroll
      for (int q = 0; q < 8; ++q) dst[q] = z;
    } else {
      const F4* src = reinterpret_cast<const F4*>(s1 + ((size_t)(b * RB + rin) * AB + ain) * C1);
#pragma unroll
      for (int q = 0; q < 8; ++q) dst[q] = src[q];
    }
  }
  __syncthreads();

  int wave = tid >> 5, lane = tid & 31;
  int lm = lane & 15, hi = lane >> 4;
  int abase = wave * 16;

  v8f acc[8] = {};
  __builtin_prefetch(w2p, 0, 3);   // global_prefetch_b8: weights stay hot in L0/L2

#pragma unroll
  for (int tap = 0; tap < 9; ++tap) {
    int da = tap % 3;
    int ai = abase + lm + da;      // tile a-index (already +1 shifted for padding)
#pragma unroll
    for (int kc = 0; kc < 2; ++kc) {
      // A fragment: lane holds M=lm; hi=0 -> K {0..7,16..23}, hi=1 -> K {8..15,24..31}
      v16h afrag;
      const _Float16* ap = &tile[tap / 3][ai][kc * 32 + hi * 8];
      reinterpret_cast<F4*>(&afrag)[0] = *reinterpret_cast<const F4*>(ap);
      reinterpret_cast<F4*>(&afrag)[1] = *reinterpret_cast<const F4*>(ap + 16);

      const _Float16* wb = w2p + (size_t)((tap * 2 + kc) * 8) * 512 + lane * 16;
#pragma unroll
      for (int nt = 0; nt < 8; ++nt) {
        v16h bfrag;
        const _Float16* bp = wb + (size_t)nt * 512;
        reinterpret_cast<F4*>(&bfrag)[0] = *reinterpret_cast<const F4*>(bp);
        reinterpret_cast<F4*>(&bfrag)[1] = *reinterpret_cast<const F4*>(bp + 8);
        acc[nt] = __builtin_amdgcn_wmma_f32_16x16x32_f16(
            false, afrag, false, bfrag, (short)0, acc[nt], false, false);
      }
    }
  }

  // Epilogue: bias + LIF soft-reset in fp32, stream v2, reduce spike stats.
  float localspikes = 0.f;
#pragma unroll
  for (int nt = 0; nt < 8; ++nt) {
    int co = nt * 16 + lm;
    float bias = b2[co];
    float colsum = 0.f;
#pragma unroll
    for (int i = 0; i < 8; ++i) {
      int a = abase + hi * 8 + i;  // D layout: VGPR i -> M = i (+8 for hi lanes)
      size_t idx = ((size_t)blockIdx.x * AB + a) * C2 + co;
      float conv = acc[nt][i] + bias;
      float v = 0.9f * v2[idx] + conv;
      float s = (v > 1.0f) ? 1.0f : 0.0f;
      v2[idx] = v - s;
      colsum += s;
    }
    localspikes += colsum;
    atomicAdd(&s2sum[(size_t)t * (B_SZ * C2) + b * C2 + co], colsum);
  }
  atomicAdd(&sred, localspikes);
  __syncthreads();
  if (tid == 0) atomicAdd(cnt + 1, sred);
}

// ---------------------------------------------------------------------------
// Readout: logits[t] = mean_s2 @ wh + bh ; accumulate readout mean over T.
// ---------------------------------------------------------------------------
__global__ __launch_bounds__(256) void readout_kernel(const float* s2sum, const float* wh,
                                                      const float* bh, float* out, int t) {
  int i = blockIdx.x * 256 + threadIdx.x;
  if (i >= B_SZ * OUT_DIM) return;
  int b = i / OUT_DIM, o = i % OUT_DIM;
  const float* sv = s2sum + (size_t)t * (B_SZ * C2) + b * C2;
  const float inv = 1.0f / (RB * AB);
  float y = bh[o];
#pragma unroll 4
  for (int co = 0; co < C2; ++co) y += sv[co] * inv * wh[(size_t)co * OUT_DIM + o];
  out[OFF_LOGITS + (size_t)t * (B_SZ * OUT_DIM) + i] = y;
  out[OFF_READOUT + i] += y * (1.0f / T_STEPS);
}

__global__ void finalize_kernel(const float* cnt, float* out) {
  if (threadIdx.x == 0) out[OFF_SR + 0] = cnt[0] * (1.0f / 67108864.0f);  // T*B*RB*AB*C1
  if (threadIdx.x == 1) out[OFF_SR + 1] = cnt[1] * (1.0f / 134217728.0f); // T*B*RB*AB*C2
}

// ---------------------------------------------------------------------------
extern "C" void kernel_launch(void* const* d_in, const int* in_sizes, int n_in,
                              void* d_out, int out_size, void* d_ws, size_t ws_size,
                              hipStream_t stream) {
  (void)in_sizes; (void)n_in; (void)out_size; (void)ws_size;
  const float* x_seq = (const float*)d_in[0];
  const float* w1    = (const float*)d_in[1];
  const float* b1    = (const float*)d_in[2];
  const float* w2    = (const float*)d_in[3];
  const float* b2    = (const float*)d_in[4];
  const float* wh    = (const float*)d_in[5];
  const float* bh    = (const float*)d_in[6];
  float* out = (float*)d_out;

  char* ws = (char*)d_ws;
  float*    v1    = (float*)(ws + WS_V1);
  float*    v2    = (float*)(ws + WS_V2);
  _Float16* s1    = (_Float16*)(ws + WS_S1);
  _Float16* w2p   = (_Float16*)(ws + WS_W2P);
  float*    s2sum = (float*)(ws + WS_S2SUM);
  float*    cnt   = (float*)(ws + WS_CNT);

  init_kernel<<<32768, 256, 0, stream>>>(v1, v2, s2sum, cnt, out);
  pack_w2_kernel<<<288, 256, 0, stream>>>(w2, w2p);
  sample_kernel<<<T_STEPS * B_SZ * RB, 384, 0, stream>>>(x_seq, out);

  for (int t = 0; t < T_STEPS; ++t) {   // LIF recurrence: stream-ordered
    conv1_lif_kernel<<<(B_SZ * RB * AB) / 4, 256, 0, stream>>>(out, w1, b1, v1, s1, cnt, t);
    conv2_lif_wmma_kernel<<<B_SZ * RB, 256, 0, stream>>>(s1, w2p, b2, v2, s2sum, cnt, t);
    readout_kernel<<<(B_SZ * OUT_DIM + 255) / 256, 256, 0, stream>>>(s2sum, wh, bh, out, t);
  }
  finalize_kernel<<<1, 32, 0, stream>>>(cnt, out);
}
